// GraphEncoder_34626026341050
// MI455X (gfx1250) — matrix operand order repaired and checked by reading, not compiled, not run on hardware
//
#include <hip/hip_runtime.h>

#define S_LEN 2048
#define HDIM  64
#define NHEADS 12
#define NBATCH 2
#define RBUCK 256
#define HID   768
#define MTOK  4096  // NBATCH*S_LEN

typedef __bf16 bf16_t;
typedef __attribute__((ext_vector_type(16))) __bf16 v16bf;
typedef __attribute__((ext_vector_type(8)))  __bf16 v8bf;
typedef __attribute__((ext_vector_type(8)))  float  v8f;

static __device__ __forceinline__ v8f wmma_bf16(v16bf a, v16bf b, v8f c) {
  // D = A(16x32 bf16) * B(32x16 bf16) + C(16x16 f32)
  return __builtin_amdgcn_wmma_f32_16x16x32_bf16(false, a, false, b, (short)0, c,
                                                 false, false);
}

// A-fragment (16x32 bf16): lane r=lane&15 holds row r.
// elements e -> K = 16*hi + e
static __device__ __forceinline__ v16bf load_a(const bf16_t* row, int hi) {
  v8bf lo = *(const v8bf*)(row + 16 * hi);
  v8bf hh = *(const v8bf*)(row + 16 * hi + 8);
  v16bf r;
#pragma unroll
  for (int e = 0; e < 8; ++e) { r[e] = lo[e]; r[8 + e] = hh[e]; }
  return r;
}

// ---------------- pack kernels ----------------
__global__ void cvt_bf16_kernel(const float* __restrict__ src,
                                bf16_t* __restrict__ dst, int n) {
  int i = blockIdx.x * 256 + threadIdx.x;
  if (i < n) dst[i] = (bf16_t)src[i];
}

__global__ void transpose_bf16_kernel(const float* __restrict__ src,
                                      bf16_t* __restrict__ dst, int rows,
                                      int cols) {
  int i = blockIdx.x * 256 + threadIdx.x;
  if (i < rows * cols) {
    int r = i / cols, c = i - r * cols;
    dst[(size_t)c * rows + r] = (bf16_t)src[i];
  }
}

// ---------------- QKV projection: Y = X@W + b ----------------
// per wave: 16 tokens x 64 features (= one head) of one of {q,k,v}
__global__ __launch_bounds__(128) void proj_kernel(
    const bf16_t* __restrict__ X,    // [4096,768] bf16
    const bf16_t* __restrict__ Wt3,  // 3 x [768n][768k] bf16 (W transposed)
    const float* __restrict__ bq, const float* __restrict__ bk,
    const float* __restrict__ bv, bf16_t* __restrict__ qg,
    bf16_t* __restrict__ kg, bf16_t* __restrict__ vtg) {
  const int lane = threadIdx.x & 31;
  const int gw = blockIdx.x * 4 + (threadIdx.x >> 5);
  const int h = gw % NHEADS;
  const int t = gw / NHEADS;
  const int mt = t & 255;
  const int w = t >> 8;  // 0=q 1=k 2=v
  const int m0 = mt * 16, n0 = h * HDIM;
  const int c = lane & 15, hi = (lane >> 4) & 1;
  const bf16_t* W = Wt3 + (size_t)w * HID * HID;
  const float* bias = (w == 0) ? bq : ((w == 1) ? bk : bv);

  v8f acc[4];
#pragma unroll
  for (int nt = 0; nt < 4; ++nt) {
    float bb = bias[n0 + nt * 16 + c];
#pragma unroll
    for (int p = 0; p < 8; ++p) acc[nt][p] = bb;
  }
  for (int kk = 0; kk < HID / 32; ++kk) {
    int k0 = kk * 32;
    v16bf ax = load_a(X + (size_t)(m0 + c) * HID + k0, hi);
#pragma unroll
    for (int nt = 0; nt < 4; ++nt)
      acc[nt] = wmma_bf16(
          ax, *(const v16bf*)(W + (size_t)(n0 + nt * 16 + c) * HID + k0 + 16 * hi),
          acc[nt]);
  }
  const int b = m0 >> 11;           // token / 2048
  const int sb = m0 & (S_LEN - 1);  // token within batch
  const size_t head = (size_t)(b * NHEADS + h);
  if (w < 2) {  // q,k row-major [S,64]
    bf16_t* dst = (w == 0 ? qg : kg) + head * S_LEN * HDIM;
#pragma unroll
    for (int nt = 0; nt < 4; ++nt)
#pragma unroll
      for (int p = 0; p < 8; ++p)
        dst[(size_t)(sb + p + 8 * hi) * HDIM + nt * 16 + c] = (bf16_t)acc[nt][p];
  } else {  // v transposed [64,S]
    bf16_t* dst = vtg + head * HDIM * S_LEN;
#pragma unroll
    for (int nt = 0; nt < 4; ++nt) {
      v8bf pk;
#pragma unroll
      for (int p = 0; p < 8; ++p) pk[p] = (bf16_t)acc[nt][p];
      *(v8bf*)(dst + (size_t)(nt * 16 + c) * S_LEN + sb + 8 * hi) = pk;
    }
  }
}

// ---------------- rel_scores = q @ rel_key_emb^T  (raw, unscaled) ----------
__global__ __launch_bounds__(128) void relsc_kernel(
    const bf16_t* __restrict__ qg, const bf16_t* __restrict__ rke,
    float* __restrict__ rels) {
  const int lane = threadIdx.x & 31;
  const int gw = blockIdx.x * 4 + (threadIdx.x >> 5);
  const int rt = gw & 3;
  const int mt = (gw >> 2) & 127;
  const int bh = gw >> 9;
  const int m0 = mt * 16, r0 = rt * 64;
  const int c = lane & 15, hi = (lane >> 4) & 1;
  const bf16_t* qrow = qg + (size_t)bh * S_LEN * HDIM + (size_t)(m0 + c) * HDIM;
  v16bf a0 = load_a(qrow, hi);
  v16bf a1 = load_a(qrow + 32, hi);
  v8f acc[4];
#pragma unroll
  for (int nt = 0; nt < 4; ++nt) {
#pragma unroll
    for (int p = 0; p < 8; ++p) acc[nt][p] = 0.f;
    const bf16_t* bcol = rke + (size_t)(r0 + nt * 16 + c) * HDIM;
    acc[nt] = wmma_bf16(a0, *(const v16bf*)(bcol + 16 * hi), acc[nt]);
    acc[nt] = wmma_bf16(a1, *(const v16bf*)(bcol + 32 + 16 * hi), acc[nt]);
  }
  float* dst = rels + (size_t)bh * S_LEN * RBUCK;
#pragma unroll
  for (int nt = 0; nt < 4; ++nt)
#pragma unroll
    for (int p = 0; p < 8; ++p)
      dst[(size_t)(m0 + p + 8 * hi) * RBUCK + r0 + nt * 16 + c] = acc[nt][p];
}

// ---------------- fused flash attention w/ relative position ----------------
__global__ __launch_bounds__(64) void attn_kernel(
    const bf16_t* __restrict__ qg, const bf16_t* __restrict__ kg,
    const bf16_t* __restrict__ vtg, const float* __restrict__ rels,
    const bf16_t* __restrict__ rveT, const float* __restrict__ maskp,
    float* __restrict__ out) {
  __shared__ __align__(32) float relp_s[2][16 * RBUCK];   // per-wave rel_probs
  __shared__ __align__(32) bf16_t psc_s[2][16 * 32];      // P transpose staging

  const int lane = threadIdx.x & 31;
  const int wid = threadIdx.x >> 5;
  const int gw = blockIdx.x * 2 + wid;
  const int mt = gw & 127;
  const int bh = gw >> 7;
  const int b = bh / NHEADS;
  const int h = bh - b * NHEADS;
  const int i0 = mt * 16;
  const int c = lane & 15;
  const int hi = (lane >> 4) & 1;
  const float scale = 0.125f;  // 1/sqrt(64)

  const bf16_t* kbh = kg + (size_t)bh * S_LEN * HDIM;
  const bf16_t* vbh = vtg + (size_t)bh * HDIM * S_LEN;
  const float* rbh = rels + (size_t)bh * S_LEN * RBUCK;
  const float* mk = maskp + (size_t)b * S_LEN;

  const bf16_t* qrow =
      qg + (size_t)bh * S_LEN * HDIM + (size_t)(i0 + c) * HDIM;
  v16bf aq0 = load_a(qrow, hi);
  v16bf aq1 = load_a(qrow + 32, hi);

  v8f acc[4];
#pragma unroll
  for (int nt = 0; nt < 4; ++nt)
#pragma unroll
    for (int p = 0; p < 8; ++p) acc[nt][p] = 0.f;

  float mrow[8], lrow[8];
#pragma unroll
  for (int p = 0; p < 8; ++p) { mrow[p] = -3.0e38f; lrow[p] = 0.f; }

  float* relp = &relp_s[wid][0];
  bf16_t* psc = &psc_s[wid][0];

  // scores for a 16x32 tile: (q.k + rel_bias)*scale + mask_bias
  auto calc_tile = [&](int j0, float f0[8], float f1[8]) {
    const bf16_t* kr0 = kbh + (size_t)(j0 + c) * HDIM;
    const bf16_t* kr1 = kbh + (size_t)(j0 + 16 + c) * HDIM;
    v8f s0 = {}, s1 = {};
    s0 = wmma_bf16(aq0, *(const v16bf*)(kr0 + 16 * hi), s0);
    s0 = wmma_bf16(aq1, *(const v16bf*)(kr0 + 32 + 16 * hi), s0);
    s1 = wmma_bf16(aq0, *(const v16bf*)(kr1 + 16 * hi), s1);
    s1 = wmma_bf16(aq1, *(const v16bf*)(kr1 + 32 + 16 * hi), s1);
    int j0c = j0 + c, j1c = j0 + 16 + c;
    float mb0 = (1.f - mk[j0c]) * -10000.f;
    float mb1 = (1.f - mk[j1c]) * -10000.f;
#pragma unroll
    for (int p = 0; p < 8; ++p) {
      int irow = i0 + p + 8 * hi;
      const float* rr = rbh + (size_t)irow * RBUCK;
      int r0i = min(max(j0c - irow + 128, 0), 255);
      int r1i = min(max(j1c - irow + 128, 0), 255);
      f0[p] = (s0[p] + rr[r0i]) * scale + mb0;
      f1[p] = (s1[p] + rr[r1i]) * scale + mb1;
    }
  };

  // ---- Phase 1: flash softmax + ctx = P @ V ----
  for (int jt = 0; jt < S_LEN / 32; ++jt) {
    int j0 = jt * 32;
    float f0[8], f1[8], alpha[8];
    calc_tile(j0, f0, f1);
#pragma unroll
    for (int p = 0; p < 8; ++p) {
      float tm = fmaxf(f0[p], f1[p]);
      for (int off = 1; off < 16; off <<= 1)
        tm = fmaxf(tm, __shfl_xor(tm, off, 32));
      float mn = fmaxf(mrow[p], tm);
      alpha[p] = __expf(mrow[p] - mn);
      mrow[p] = mn;
      f0[p] = __expf(f0[p] - mn);
      f1[p] = __expf(f1[p] - mn);
      float rs = f0[p] + f1[p];
      for (int off = 1; off < 16; off <<= 1) rs += __shfl_xor(rs, off, 32);
      lrow[p] = lrow[p] * alpha[p] + rs;
    }
    // C-layout -> A-layout transpose via per-wave LDS
#pragma unroll
    for (int p = 0; p < 8; ++p) {
      psc[(p + 8 * hi) * 32 + c] = (bf16_t)f0[p];
      psc[(p + 8 * hi) * 32 + 16 + c] = (bf16_t)f1[p];
    }
    asm volatile("s_wait_dscnt 0" ::: "memory");
    v16bf pa = load_a(psc + c * 32, hi);
    asm volatile("s_wait_dscnt 0" ::: "memory");
#pragma unroll
    for (int nt = 0; nt < 4; ++nt) {
#pragma unroll
      for (int p = 0; p < 8; ++p) acc[nt][p] *= alpha[p];
      acc[nt] = wmma_bf16(
          pa,
          *(const v16bf*)(vbh + (size_t)(nt * 16 + c) * S_LEN + j0 + 16 * hi),
          acc[nt]);
    }
  }

  // ---- Phase 2: scatter unnormalized probs into rel buckets ----
  for (int t = lane; t < 16 * RBUCK; t += 32) relp[t] = 0.f;
  asm volatile("s_wait_dscnt 0" ::: "memory");
  for (int jt = 0; jt < S_LEN / 32; ++jt) {
    int j0 = jt * 32;
    float f0[8], f1[8];
    calc_tile(j0, f0, f1);
    int j0c = j0 + c, j1c = j0 + 16 + c;
#pragma unroll
    for (int p = 0; p < 8; ++p) {
      int irow = i0 + p + 8 * hi;
      int r0i = min(max(j0c - irow + 128, 0), 255);
      int r1i = min(max(j1c - irow + 128, 0), 255);
      atomicAdd(&relp[(p + 8 * hi) * RBUCK + r0i], __expf(f0[p] - mrow[p]));
      atomicAdd(&relp[(p + 8 * hi) * RBUCK + r1i], __expf(f1[p] - mrow[p]));
    }
  }
  asm volatile("s_wait_dscnt 0" ::: "memory");

  // ---- rel_probs(16x256) @ rel_value_emb(256x64), K-steps of 32 ----
  for (int kk = 0; kk < 8; ++kk) {
    int k0 = kk * 32;
    const float* pr = relp + c * RBUCK + k0;
    v16bf ra;
#pragma unroll
    for (int e = 0; e < 8; ++e) {
      ra[e] = (bf16_t)pr[16 * hi + e];
      ra[8 + e] = (bf16_t)pr[16 * hi + 8 + e];
    }
#pragma unroll
    for (int nt = 0; nt < 4; ++nt)
      acc[nt] = wmma_bf16(
          ra, *(const v16bf*)(rveT + (size_t)(nt * 16 + c) * RBUCK + k0 + 16 * hi),
          acc[nt]);
  }

  // ---- normalize & store [B,S,768] ----
#pragma unroll
  for (int p = 0; p < 8; ++p) {
    int irow = i0 + p + 8 * hi;
    float inv = 1.f / lrow[p];
    size_t o = ((size_t)(b * S_LEN + irow)) * HID + (size_t)h * HDIM;
#pragma unroll
    for (int nt = 0; nt < 4; ++nt) out[o + nt * 16 + c] = acc[nt][p] * inv;
  }
}

// ---------------- host launch ----------------
extern "C" void kernel_launch(void* const* d_in, const int* in_sizes, int n_in,
                              void* d_out, int out_size, void* d_ws,
                              size_t ws_size, hipStream_t stream) {
  const float* hidden = (const float*)d_in[0];
  const float* mask = (const float*)d_in[1];
  const float* Wq = (const float*)d_in[2];
  const float* bq = (const float*)d_in[3];
  const float* Wk = (const float*)d_in[4];
  const float* bk = (const float*)d_in[5];
  const float* Wv = (const float*)d_in[6];
  const float* bv = (const float*)d_in[7];
  const float* rke = (const float*)d_in[8];
  const float* rve = (const float*)d_in[9];
  float* out = (float*)d_out;

  char* p = (char*)d_ws;
  auto alloc = [&](size_t bytes) -> char* {
    char* r = p;
    p += (bytes + 255) & ~(size_t)255;
    return r;
  };
  bf16_t* Xbf = (bf16_t*)alloc((size_t)MTOK * HID * 2);
  bf16_t* Wt3 = (bf16_t*)alloc((size_t)3 * HID * HID * 2);
  bf16_t* rkeb = (bf16_t*)alloc((size_t)RBUCK * HDIM * 2);
  bf16_t* rveT = (bf16_t*)alloc((size_t)HDIM * RBUCK * 2);
  bf16_t* qg = (bf16_t*)alloc((size_t)NBATCH * NHEADS * S_LEN * HDIM * 2);
  bf16_t* kg = (bf16_t*)alloc((size_t)NBATCH * NHEADS * S_LEN * HDIM * 2);
  bf16_t* vtg = (bf16_t*)alloc((size_t)NBATCH * NHEADS * HDIM * S_LEN * 2);
  float* rels = (float*)alloc((size_t)NBATCH * NHEADS * S_LEN * RBUCK * 4);

  // pack / convert
  {
    int n = MTOK * HID;
    cvt_bf16_kernel<<<(n + 255) / 256, 256, 0, stream>>>(hidden, Xbf, n);
  }
  {
    int n = HID * HID;
    transpose_bf16_kernel<<<(n + 255) / 256, 256, 0, stream>>>(Wq, Wt3, HID, HID);
    transpose_bf16_kernel<<<(n + 255) / 256, 256, 0, stream>>>(
        Wk, Wt3 + (size_t)HID * HID, HID, HID);
    transpose_bf16_kernel<<<(n + 255) / 256, 256, 0, stream>>>(
        Wv, Wt3 + (size_t)2 * HID * HID, HID, HID);
  }
  {
    int n = RBUCK * HDIM;
    cvt_bf16_kernel<<<(n + 255) / 256, 256, 0, stream>>>(rke, rkeb, n);
    transpose_bf16_kernel<<<(n + 255) / 256, 256, 0, stream>>>(rve, rveT, RBUCK,
                                                               HDIM);
  }

  // QKV projections: 3 * 256 m-tiles * 12 heads = 9216 waves, 4 waves/block
  proj_kernel<<<9216 / 4, 128, 0, stream>>>(Xbf, Wt3, bq, bk, bv, qg, kg, vtg);

  // rel_scores: 24 heads * 128 m-tiles * 4 r-strips = 12288 waves
  relsc_kernel<<<12288 / 4, 128, 0, stream>>>(qg, rkeb, rels);

  // attention: 24 heads * 128 m-tiles = 3072 waves, 2 waves/block
  attn_kernel<<<3072 / 2, 64, 0, stream>>>(qg, kg, vtg, rels, rveT, mask, out);
}
